// EvaAttention_58712202936928
// MI455X (gfx1250) — matrix operand order repaired and checked by reading, not compiled, not run on hardware
//
#include <hip/hip_runtime.h>
#include <hip/hip_bf16.h>

// ---------------------------------------------------------------------------
// EVA attention for MI455X (gfx1250, wave32, WMMA).
// f16 inputs to v_wmma_f32_16x16x32_f16, f32 accumulation.
// V is stored transposed so every WMMA B-fragment is one contiguous 32B load.
// ---------------------------------------------------------------------------

typedef __attribute__((ext_vector_type(16))) _Float16 v16h;
typedef __attribute__((ext_vector_type(8)))  _Float16 v8h;
typedef __attribute__((ext_vector_type(8)))  float    v8f;

union Frag16 { v16h v; v8h h[2]; };

#define SEQ_N 2048
#define DIM_C 1024
#define HEADS 16
#define HDIM  64

// ---------------- prep kernels ----------------

__global__ __launch_bounds__(256) void to_f16_kernel(const float* __restrict__ s,
                                                     _Float16* __restrict__ d, int n) {
    int i = blockIdx.x * blockDim.x + threadIdx.x;
    if (i < n) d[i] = (_Float16)s[i];
}

// dst[c][r] = (f16) src[r][c];  src is R x Ccol row-major, dst is Ccol x R row-major.
__global__ __launch_bounds__(256) void transpose_f16_kernel(const float* __restrict__ src,
                                                            _Float16* __restrict__ dst,
                                                            int R, int Ccol) {
    int i = blockIdx.x * blockDim.x + threadIdx.x;
    if (i >= R * Ccol) return;
    int cd = i / R;        // dst row  (= src col)
    int rd = i % R;        // dst col  (= src row)
    dst[i] = (_Float16)src[rd * Ccol + cd];
}

// idx[n] = clip(cumsum(!cm)[n] - 1, 0, Mrope-1).  cm is a byte-bool array.
__global__ __launch_bounds__(256) void build_idx_kernel(const unsigned char* __restrict__ cm,
                                                        int* __restrict__ idx, int n, int Mrope) {
    int i = blockIdx.x * blockDim.x + threadIdx.x;
    if (i >= n) return;
    int cnt = 0;
    for (int j = 0; j <= i; ++j) cnt += cm[j] ? 0 : 1;
    int v = cnt - 1;
    v = v < 0 ? 0 : v;
    v = v > Mrope - 1 ? Mrope - 1 : v;
    idx[i] = v;
}

// cosT/sinT: [SEQ_N][HDIM] f32; class positions get cos=1, sin=0.
__global__ __launch_bounds__(256) void build_sincos_kernel(const unsigned char* __restrict__ cm,
                                                           const int* __restrict__ idx,
                                                           const float* __restrict__ rope,
                                                           float* __restrict__ cosT,
                                                           float* __restrict__ sinT,
                                                           int n, int d) {
    int t = blockIdx.x * blockDim.x + threadIdx.x;
    if (t >= n * d) return;
    int i = t / d, dd = t % d;
    if (cm[i]) { cosT[t] = 1.0f; sinT[t] = 0.0f; }
    else {
        int r = idx[i];
        sinT[t] = rope[r * 2 * d + dd];       // sin half
        cosT[t] = rope[r * 2 * d + d + dd];   // cos half
    }
}

// ---------------- WMMA GEMM ----------------
// C[M x Nn] = A[M x K] * B, with B supplied transposed (Bt[Nn x K] row-major) so
// every B-fragment is one contiguous 32B load per lane.
// MODE 0: out = acc + bias                  -> f32 [M][Nn]   (final projection)
// MODE 1: out = acc + bias                  -> f16 [Nn][M]   (V, stored transposed)
// MODE 2: out = rope(acc[+bias])*scale      -> f16 [M][Nn]   (Q: bias+scale, K: bias=null)
template <int MODE>
__global__ __launch_bounds__(256) void gemm_wmma(const _Float16* __restrict__ A,
                                                 const _Float16* __restrict__ Bt,
                                                 const float* __restrict__ bias,
                                                 float* __restrict__ outF,
                                                 _Float16* __restrict__ outH,
                                                 const float* __restrict__ cosT,
                                                 const float* __restrict__ sinT,
                                                 int M, int Nn, int K, float scale) {
    const int lane = threadIdx.x & 31;
    const int wave = threadIdx.x >> 5;
    const int ntiles = Nn >> 6;                       // 64-wide N tiles per wave
    const int gw = blockIdx.x * (blockDim.x >> 5) + wave;
    if (gw >= (M >> 4) * ntiles) return;              // wave-uniform guard
    const int mt = gw / ntiles, nt = gw % ntiles;
    const int m0 = mt << 4, n0 = nt << 6;
    const int half = lane >> 4, l16 = lane & 15;

    v8f acc[4];
#pragma unroll
    for (int t = 0; t < 4; ++t)
#pragma unroll
        for (int r = 0; r < 8; ++r) acc[t][r] = 0.0f;

    const int arow = m0 + l16;
    const int akb = half * 16;    // A frag: halves 0..15 = K kb..kb+15
    const int bkb = half * 16;    // B frag: halves 0..15 = K kb2..kb2+15, N = l16

    for (int kk = 0; kk < K; kk += 32) {
        Frag16 a;
        a.h[0] = *(const v8h*)(A + (size_t)arow * K + kk + akb);
        a.h[1] = *(const v8h*)(A + (size_t)arow * K + kk + akb + 8);
#pragma unroll
        for (int t = 0; t < 4; ++t) {
            v16h b = *(const v16h*)(Bt + (size_t)(n0 + t * 16 + l16) * K + kk + bkb);
            acc[t] = __builtin_amdgcn_wmma_f32_16x16x32_f16(
                false, a.v, false, b, (short)0, acc[t], false, false);
        }
    }

#pragma unroll
    for (int t = 0; t < 4; ++t) {
        const int c = n0 + t * 16 + l16;
        float bv = 0.0f;
        if (MODE != 2) bv = bias[c];
        else if (bias) bv = bias[c];
#pragma unroll
        for (int r = 0; r < 8; ++r) {
            const int m = m0 + r + half * 8;
            float val = acc[t][r] + bv;
            if (MODE == 0) {
                outF[(size_t)m * Nn + c] = val;
            } else if (MODE == 1) {
                outH[(size_t)c * M + m] = (_Float16)val;          // transposed store
            } else {
                // RoPE: rot(x)[even c] = -x[c+1], rot(x)[odd c] = x[c-1].
                // Paired column lives in lane^1 of the same 16-lane group.
                float partner = __shfl_xor(val, 1, 32);
                float rotv = (c & 1) ? partner : -partner;
                int dd = c & (HDIM - 1);
                float cv = cosT[m * HDIM + dd];
                float sv = sinT[m * HDIM + dd];
                outH[(size_t)m * Nn + c] = (_Float16)((val * cv + rotv * sv) * scale);
            }
        }
    }
}

// ---------------- flash attention ----------------

__device__ __forceinline__ float red_max16(float v) {
#pragma unroll
    for (int m = 1; m < 16; m <<= 1) v = fmaxf(v, __shfl_xor(v, m, 32));
    return v;
}
__device__ __forceinline__ float red_sum16(float v) {
#pragma unroll
    for (int m = 1; m < 16; m <<= 1) v += __shfl_xor(v, m, 32);
    return v;
}

// One wave = one (head, 16-query tile). Online softmax over 64 key-blocks of 32.
// Q has 1/sqrt(d) pre-folded (exact: 0.125 is a power of two).
// Vt is [DIM_C][SEQ_N] (per-column key-major), so P@V B-fragments are contiguous.
__global__ __launch_bounds__(128) void attn_flash(const _Float16* __restrict__ Q,
                                                  const _Float16* __restrict__ Km,
                                                  const _Float16* __restrict__ Vt,
                                                  _Float16* __restrict__ O) {
    __shared__ _Float16 plds[4][16 * 32];   // per-wave P staging (C-layout -> A-layout)
    const int lane = threadIdx.x & 31, wid = threadIdx.x >> 5;
    const int gw = blockIdx.x * 4 + wid;
    const int head = gw >> 7;               // 128 query tiles per head
    const int mt = gw & 127;
    const int half = lane >> 4, l16 = lane & 15;
    const int akb = half * 16, bkb = half * 16;

    // Q fragments for K-chunks 0 and 32 (A-matrix 16x32, contiguous k order: lane half h holds k 16h..16h+15)
    const int qrow = mt * 16 + l16;
    Frag16 qf[2];
#pragma unroll
    for (int kc = 0; kc < 2; ++kc) {
        qf[kc].h[0] = *(const v8h*)(Q + (size_t)qrow * DIM_C + head * HDIM + kc * 32 + akb);
        qf[kc].h[1] = *(const v8h*)(Q + (size_t)qrow * DIM_C + head * HDIM + kc * 32 + akb + 8);
    }

    float mrow[8], lrow[8];
    v8f o[4];
#pragma unroll
    for (int r = 0; r < 8; ++r) { mrow[r] = -1e30f; lrow[r] = 0.0f; }
#pragma unroll
    for (int t = 0; t < 4; ++t)
#pragma unroll
        for (int r = 0; r < 8; ++r) o[t][r] = 0.0f;

    for (int j = 0; j < SEQ_N / 32; ++j) {
        const int j0 = j * 32;
        // S = Q * K^T for 32 keys (2 N-tiles x 2 K-chunks)
        v8f s0, s1;
#pragma unroll
        for (int r = 0; r < 8; ++r) { s0[r] = 0.0f; s1[r] = 0.0f; }
#pragma unroll
        for (int kc = 0; kc < 2; ++kc) {
            const _Float16* kp = Km + (size_t)(j0 + l16) * DIM_C + head * HDIM + kc * 32 + bkb;
            v16h b0 = *(const v16h*)kp;
            v16h b1 = *(const v16h*)(kp + (size_t)16 * DIM_C);
            s0 = __builtin_amdgcn_wmma_f32_16x16x32_f16(false, qf[kc].v, false, b0, (short)0, s0, false, false);
            s1 = __builtin_amdgcn_wmma_f32_16x16x32_f16(false, qf[kc].v, false, b1, (short)0, s1, false, false);
        }
        // online softmax; rows of a C-tile live inside one 16-lane group
#pragma unroll
        for (int r = 0; r < 8; ++r) {
            float mx = red_max16(fmaxf(s0[r], s1[r]));
            float mn = fmaxf(mrow[r], mx);
            float f = __expf(mrow[r] - mn);
            mrow[r] = mn;
            lrow[r] *= f;
            o[0][r] *= f; o[1][r] *= f; o[2][r] *= f; o[3][r] *= f;
            float p0 = __expf(s0[r] - mn);
            float p1 = __expf(s1[r] - mn);
            lrow[r] += red_sum16(p0 + p1);
            const int prow = r + half * 8;
            plds[wid][prow * 32 + l16] = (_Float16)p0;
            plds[wid][prow * 32 + 16 + l16] = (_Float16)p1;
        }
        // re-read P in A-fragment layout (intra-wave LDS ops are in-order)
        Frag16 pa;
#pragma unroll
        for (int i = 0; i < 8; ++i) {
            pa.v[i]     = plds[wid][l16 * 32 + akb + i];
            pa.v[8 + i] = plds[wid][l16 * 32 + akb + 8 + i];
        }
        // O += P (16x32) * V (32x64); Vt rows are features, cols are keys -> contiguous
#pragma unroll
        for (int t = 0; t < 4; ++t) {
            v16h bv = *(const v16h*)(Vt + (size_t)(head * HDIM + t * 16 + l16) * SEQ_N + j0 + bkb);
            o[t] = __builtin_amdgcn_wmma_f32_16x16x32_f16(false, pa.v, false, bv, (short)0, o[t], false, false);
        }
    }

#pragma unroll
    for (int t = 0; t < 4; ++t)
#pragma unroll
        for (int r = 0; r < 8; ++r) {
            const int row = mt * 16 + r + half * 8;
            O[(size_t)row * DIM_C + head * HDIM + t * 16 + l16] = (_Float16)(o[t][r] / lrow[r]);
        }
}

// ---------------- host launch ----------------

extern "C" void kernel_launch(void* const* d_in, const int* in_sizes, int n_in,
                              void* d_out, int out_size, void* d_ws, size_t ws_size,
                              hipStream_t stream) {
    const float* x      = (const float*)d_in[0];
    const float* rope   = (const float*)d_in[1];
    const unsigned char* cmask = (const unsigned char*)d_in[2]; // jnp bool -> byte
    const float* w_q    = (const float*)d_in[3];
    const float* b_q    = (const float*)d_in[4];
    const float* w_k    = (const float*)d_in[5];
    const float* w_v    = (const float*)d_in[6];
    const float* b_v    = (const float*)d_in[7];
    const float* w_proj = (const float*)d_in[8];
    const float* b_proj = (const float*)d_in[9];
    float* out = (float*)d_out;
    const int Mrope = in_sizes[1] / (2 * HDIM);

    char* ws = (char*)d_ws;
    size_t off = 0;
    auto carve = [&](size_t bytes) -> void* {
        void* p = (void*)(ws + off);
        off += (bytes + 255) & ~(size_t)255;
        return p;
    };
    _Float16* xh  = (_Float16*)carve((size_t)SEQ_N * DIM_C * 2);
    _Float16* wqt = (_Float16*)carve((size_t)DIM_C * DIM_C * 2);
    _Float16* wkt = (_Float16*)carve((size_t)DIM_C * DIM_C * 2);
    _Float16* wvt = (_Float16*)carve((size_t)DIM_C * DIM_C * 2);
    _Float16* wpt = (_Float16*)carve((size_t)DIM_C * DIM_C * 2);
    _Float16* qh  = (_Float16*)carve((size_t)SEQ_N * DIM_C * 2);
    _Float16* kh  = (_Float16*)carve((size_t)SEQ_N * DIM_C * 2);
    _Float16* vt  = (_Float16*)carve((size_t)SEQ_N * DIM_C * 2);   // V transposed [C][N]
    _Float16* oh  = (_Float16*)carve((size_t)SEQ_N * DIM_C * 2);
    float* cosT   = (float*)carve((size_t)SEQ_N * HDIM * 4);
    float* sinT   = (float*)carve((size_t)SEQ_N * HDIM * 4);
    int*   idx    = (int*)carve((size_t)SEQ_N * 4);

    to_f16_kernel<<<(SEQ_N * DIM_C + 255) / 256, 256, 0, stream>>>(x, xh, SEQ_N * DIM_C);
    const int tb = (DIM_C * DIM_C + 255) / 256;
    transpose_f16_kernel<<<tb, 256, 0, stream>>>(w_q,    wqt, DIM_C, DIM_C);
    transpose_f16_kernel<<<tb, 256, 0, stream>>>(w_k,    wkt, DIM_C, DIM_C);
    transpose_f16_kernel<<<tb, 256, 0, stream>>>(w_v,    wvt, DIM_C, DIM_C);
    transpose_f16_kernel<<<tb, 256, 0, stream>>>(w_proj, wpt, DIM_C, DIM_C);
    build_idx_kernel<<<(SEQ_N + 255) / 256, 256, 0, stream>>>(cmask, idx, SEQ_N, Mrope);
    build_sincos_kernel<<<(SEQ_N * HDIM + 255) / 256, 256, 0, stream>>>(
        cmask, idx, rope, cosT, sinT, SEQ_N, HDIM);

    const int gemm_blocks = (SEQ_N / 16) * (DIM_C / 64) / 8;   // 8 waves per block
    // Q: bias + rope + 1/sqrt(d); K: rope only; V: bias only, transposed output
    gemm_wmma<2><<<gemm_blocks, 256, 0, stream>>>(xh, wqt, b_q, nullptr, qh, cosT, sinT,
                                                  SEQ_N, DIM_C, DIM_C, 0.125f);
    gemm_wmma<2><<<gemm_blocks, 256, 0, stream>>>(xh, wkt, nullptr, nullptr, kh, cosT, sinT,
                                                  SEQ_N, DIM_C, DIM_C, 1.0f);
    gemm_wmma<1><<<gemm_blocks, 256, 0, stream>>>(xh, wvt, b_v, nullptr, vt, nullptr, nullptr,
                                                  SEQ_N, DIM_C, DIM_C, 1.0f);

    attn_flash<<<(HEADS * (SEQ_N / 16)) / 4, 128, 0, stream>>>(qh, kh, vt, oh);

    gemm_wmma<0><<<gemm_blocks, 256, 0, stream>>>(oh, wpt, b_proj, out, nullptr, nullptr, nullptr,
                                                  SEQ_N, DIM_C, DIM_C, 1.0f);
    (void)n_in; (void)out_size; (void)ws_size;
}
